// Mixer_9577777070326
// MI455X (gfx1250) — matrix operand order repaired and checked by reading, not compiled, not run on hardware
//
#include <hip/hip_runtime.h>
#include <hip/hip_bf16.h>
#include <math.h>

typedef __attribute__((ext_vector_type(16))) __bf16 v16bf;
typedef __attribute__((ext_vector_type(8)))  float  v8f;

#define BATCH 8
#define CCH   192
#define HH    64
#define WW2   64
#define NPIX  4096
#define NC4   48
#define NC8   24
#define NCD   194
#define NKEEP 2048

__device__ __forceinline__ float lrelu(float x){ return x > 0.f ? x : 0.1f*x; }
__device__ __forceinline__ float sigm(float x){ return 1.f/(1.f+__expf(-x)); }
__device__ __forceinline__ float gelu_t(float x){
  float x3 = x*x*x;
  return 0.5f*x*(1.f + tanhf(0.7978845608028654f*(x + 0.044715f*x3)));
}

// ---------------- cond 1x1 conv + LayerNorm(channel) + lrelu -> f (B,48,N)
__global__ void __launch_bounds__(256)
k_f(const float* __restrict__ x, const float* __restrict__ win,
    const float* __restrict__ bin, const float* __restrict__ lnw,
    const float* __restrict__ lnb, float* __restrict__ f)
{
  __shared__ float wL[NC4*NCD];
  int t = threadIdx.x;
  for (int i = t; i < NC4*NCD; i += 256) wL[i] = win[i];
  __syncthreads();
  int p = blockIdx.x*256 + t;
  int b = p / NPIX, pix = p % NPIX;
  int h = pix / WW2, w = pix % WW2;
  float acc[NC4];
  #pragma unroll
  for (int o = 0; o < NC4; ++o) acc[o] = bin[o];
  const float* xb = x + (size_t)b*CCH*NPIX;
  for (int i = 0; i < NCD; ++i){
    float ci;
    if (i < CCH)       ci = xb[(size_t)i*NPIX + pix];
    else if (i == CCH) ci = -1.f + 2.f*(float)w/(float)(WW2-1);
    else               ci = -1.f + 2.f*(float)h/(float)(HH-1);
    #pragma unroll
    for (int o = 0; o < NC4; ++o) acc[o] += wL[o*NCD + i]*ci;
  }
  float u = 0.f;
  #pragma unroll
  for (int o = 0; o < NC4; ++o) u += acc[o];
  u *= (1.f/NC4);
  float s = 0.f;
  #pragma unroll
  for (int o = 0; o < NC4; ++o){ float d = acc[o]-u; s += d*d; }
  s *= (1.f/NC4);
  float r = rsqrtf(s + 1e-6f);
  #pragma unroll
  for (int o = 0; o < NC4; ++o){
    float xn = (acc[o]-u)*r;
    float y  = lnw[o]*xn + lnb[o];
    f[(size_t)b*NC4*NPIX + (size_t)o*NPIX + pix] = lrelu(y);
  }
}

// ---------------- offsets (tanh MLP *8) + token score (sigmoid of logit diff)
__global__ void __launch_bounds__(256)
k_off_score(const float* __restrict__ f,
            const float* __restrict__ w1, const float* __restrict__ b1,
            const float* __restrict__ w2, const float* __restrict__ b2,
            const float* __restrict__ mw1, const float* __restrict__ mb1,
            const float* __restrict__ mw2, const float* __restrict__ mb2,
            float* __restrict__ offsets, float* __restrict__ score)
{
  __shared__ float w1L[NC8*NC4];
  __shared__ float w2L[2*NC8];
  int t = threadIdx.x;
  for (int i = t; i < NC8*NC4; i += 256) w1L[i] = w1[i];
  if (t < 2*NC8) w2L[t] = w2[t];
  __syncthreads();
  int p = blockIdx.x*256 + t; int b = p/NPIX, pix = p%NPIX;
  const float* fb = f + (size_t)b*NC4*NPIX;
  float fv[NC4]; float tm = 0.f;
  #pragma unroll
  for (int c = 0; c < NC4; ++c){ fv[c] = fb[(size_t)c*NPIX + pix]; tm += fv[c]; }
  tm *= (1.f/NC4);
  float hbuf[NC8];
  #pragma unroll
  for (int j = 0; j < NC8; ++j){
    float a = b1[j];
    #pragma unroll
    for (int c = 0; c < NC4; ++c) a += w1L[j*NC4+c]*fv[c];
    hbuf[j] = lrelu(a);
  }
  float o0 = b2[0], o1 = b2[1];
  #pragma unroll
  for (int j = 0; j < NC8; ++j){ o0 += w2L[j]*hbuf[j]; o1 += w2L[NC8+j]*hbuf[j]; }
  offsets[(size_t)b*2*NPIX + pix]        = tanhf(o0)*8.f;
  offsets[(size_t)b*2*NPIX + NPIX + pix] = tanhf(o1)*8.f;
  float s1 = lrelu(tm*mw1[0] + mb1[0]);
  score[(size_t)b*NPIX + pix] = sigm((mw2[0]-mw2[1])*s1 + (mb2[0]-mb2[1]));
}

// ---------------- sa = sigmoid(conv3x3(f)), single output channel
__global__ void __launch_bounds__(256)
k_sa(const float* __restrict__ f, const float* __restrict__ w,
     const float* __restrict__ bia, float* __restrict__ sa)
{
  __shared__ float wL[NC4*9];
  int t = threadIdx.x;
  for (int i = t; i < NC4*9; i += 256) wL[i] = w[i];
  __syncthreads();
  int p = blockIdx.x*256 + t; int b = p/NPIX, pix = p%NPIX;
  int h = pix/WW2, wc = pix%WW2;
  const float* fb = f + (size_t)b*NC4*NPIX;
  float acc = bia[0];
  for (int c = 0; c < NC4; ++c)
    #pragma unroll
    for (int dy = -1; dy <= 1; ++dy)
      #pragma unroll
      for (int dx = -1; dx <= 1; ++dx){
        int hy = h+dy, wx = wc+dx;
        if (hy < 0 || hy >= HH || wx < 0 || wx >= WW2) continue;
        acc += wL[c*9 + (dy+1)*3 + (dx+1)] * fb[(size_t)c*NPIX + hy*WW2 + wx];
      }
  sa[(size_t)b*NPIX + pix] = sigm(acc);
}

// ---------------- f.mean over pixels -> (B,48)
__global__ void __launch_bounds__(256)
k_fmean(const float* __restrict__ f, float* __restrict__ fmean){
  int c = blockIdx.x, b = blockIdx.y, t = threadIdx.x;
  __shared__ float red[256];
  const float* src = f + ((size_t)b*NC4 + c)*NPIX;
  float s = 0.f;
  for (int p = t; p < NPIX; p += 256) s += src[p];
  red[t] = s; __syncthreads();
  for (int st = 128; st > 0; st >>= 1){ if (t < st) red[t] += red[t+st]; __syncthreads(); }
  if (t == 0) fmean[b*NC4 + c] = red[0] * (1.f/NPIX);
}

// ---------------- ca = sigmoid(fmean @ Wca^T + b)
__global__ void __launch_bounds__(256)
k_ca(const float* __restrict__ fmean, const float* __restrict__ w,
     const float* __restrict__ bia, float* __restrict__ ca){
  int i = blockIdx.x*256 + threadIdx.x;
  if (i >= BATCH*CCH) return;
  int b = i/CCH, o = i%CCH;
  float acc = bia[o];
  #pragma unroll
  for (int c = 0; c < NC4; ++c) acc += w[o*NC4+c]*fmean[b*NC4+c];
  ca[i] = sigm(acc);
}

// ---------------- stable rank (== jnp.argsort(-score) position); scatter index lists
__global__ void __launch_bounds__(256)
k_rank(const float* __restrict__ score, int* __restrict__ idx1, int* __restrict__ idx2){
  __shared__ float sc[256];
  int t = threadIdx.x;
  int b = blockIdx.x >> 4;
  int i = ((blockIdx.x & 15) << 8) + t;
  float si = score[(size_t)b*NPIX + i];
  int cnt = 0;
  for (int jt = 0; jt < NPIX; jt += 256){
    sc[t] = score[(size_t)b*NPIX + jt + t];
    __syncthreads();
    for (int jj = 0; jj < 256; ++jj){
      float sj = sc[jj]; int j = jt + jj;
      cnt += (sj > si) || (sj == si && j < i);
    }
    __syncthreads();
  }
  if (cnt < NKEEP) idx1[b*NKEEP + cnt] = i;
  else             idx2[b*NKEEP + cnt - NKEEP] = i;
}

// ---------------- generic bf16 WMMA GEMM: O[m,n] = sum_k A[m,k]*W[n,k] + bias[n]
// AMODE 0: A fp32 planar (elem at A[k*ldA+m]); AMODE 1: A bf16 row-major (A[m*ldA+k])
// OMODE 0: O fp32 planar (O[n*ldO+m]);        OMODE 1: O bf16 row-major (O[m*ldO+n])
template<int AMODE, int OMODE>
__global__ void __launch_bounds__(256)
k_gemm(const void* __restrict__ Aptr, int ldA, long strideA,
       const float* __restrict__ Wt, int Kdim,
       const float* __restrict__ bias,
       void* __restrict__ Optr, int ldO, long strideO)
{
  __shared__ __align__(16) __bf16 aL[64*32];
  __shared__ __align__(16) __bf16 bL[32*64];
  const int t = threadIdx.x, lane = t & 31, wave = t >> 5;
  const int lo = lane & 15, hi = lane >> 4;
  const int wm = wave & 3, wn = wave >> 2;
  const int mblk = blockIdx.x*64, nblk = blockIdx.y*64;
  const int b = blockIdx.z;
  v8f acc0 = {}; v8f acc1 = {};
  for (int kb = 0; kb < Kdim; kb += 32){
    if (AMODE == 0){
      const float* A = (const float*)Aptr + (size_t)b*strideA;
      #pragma unroll
      for (int i = 0; i < 8; ++i){
        int e = t + 256*i; int k = e >> 6, m = e & 63;
        aL[m*32 + k] = (__bf16)A[(size_t)(kb + k)*ldA + mblk + m];
      }
    } else {
      const __bf16* A = (const __bf16*)Aptr + (size_t)b*strideA;
      int m = t >> 2, kq = (t & 3)*8;
      *(uint4*)&aL[m*32 + kq] = *(const uint4*)&A[(size_t)(mblk + m)*ldA + kb + kq];
    }
    #pragma unroll
    for (int i = 0; i < 8; ++i){
      int e = t + 256*i; int k = e & 31, n = e >> 5;
      bL[k*64 + n] = (__bf16)Wt[(size_t)(nblk + n)*Kdim + kb + k];
    }
    if (kb + 32 < Kdim){  // warm L2 for next A tile
      if (AMODE == 0)
        __builtin_prefetch((const float*)Aptr + (size_t)b*strideA + (size_t)(kb+32)*ldA + mblk, 0, 3);
      else
        __builtin_prefetch((const __bf16*)Aptr + (size_t)b*strideA + (size_t)mblk*ldA + kb + 32, 0, 3);
    }
    __syncthreads();
    v16bf af;
    #pragma unroll
    for (int e = 0; e < 16; ++e){
      int k = e + 16*hi;
      af[e] = aL[(wm*16 + lo)*32 + k];
    }
    v16bf bf0, bf1;
    #pragma unroll
    for (int e = 0; e < 16; ++e){
      bf0[e] = bL[(e + 16*hi)*64 + wn*32 + lo];
      bf1[e] = bL[(e + 16*hi)*64 + wn*32 + 16 + lo];
    }
    acc0 = __builtin_amdgcn_wmma_f32_16x16x32_bf16(false, af, false, bf0, (short)0, acc0, false, false);
    acc1 = __builtin_amdgcn_wmma_f32_16x16x32_bf16(false, af, false, bf1, (short)0, acc1, false, false);
    __syncthreads();
  }
  int n0 = nblk + wn*32;
  float bi0 = bias[n0 + lo], bi1 = bias[n0 + 16 + lo];
  #pragma unroll
  for (int r = 0; r < 8; ++r){
    int m = mblk + wm*16 + r + 8*hi;
    float v0 = acc0[r] + bi0, v1 = acc1[r] + bi1;
    if (OMODE == 0){
      float* O = (float*)Optr + (size_t)b*strideO;
      O[(size_t)(n0 + lo)*ldO + m]      = v0;
      O[(size_t)(n0 + 16 + lo)*ldO + m] = v1;
    } else {
      __bf16* O = (__bf16*)Optr + (size_t)b*strideO;
      O[(size_t)m*ldO + n0 + lo]      = (__bf16)v0;
      O[(size_t)m*ldO + n0 + 16 + lo] = (__bf16)v1;
    }
  }
}

// ---------------- k_img = x + bilinear flow_warp(x, offsets)
__global__ void __launch_bounds__(256)
k_warp(const float* __restrict__ x, const float* __restrict__ offsets,
       float* __restrict__ kimg){
  int p = blockIdx.x*256 + threadIdx.x;
  int b = p / NPIX, pix = p % NPIX;
  int h = pix / WW2, w = pix % WW2;
  float fx = offsets[(size_t)b*2*NPIX + pix];
  float fy = offsets[(size_t)b*2*NPIX + NPIX + pix];
  float sx = (float)w + fx, sy = (float)h + fy;
  float x0f = floorf(sx), y0f = floorf(sy);
  float wx = sx - x0f, wy = sy - y0f;
  float wgt[4] = {(1.f-wx)*(1.f-wy), wx*(1.f-wy), (1.f-wx)*wy, wx*wy};
  int idx[4];
  #pragma unroll
  for (int q = 0; q < 4; ++q){
    float xi = x0f + (float)(q & 1), yi = y0f + (float)(q >> 1);
    bool valid = (xi >= 0.f) && (xi <= (float)(WW2-1)) && (yi >= 0.f) && (yi <= (float)(HH-1));
    if (!valid) wgt[q] = 0.f;
    int xc = min(max((int)xi, 0), WW2-1);
    int yc = min(max((int)yi, 0), HH-1);
    idx[q] = yc*WW2 + xc;
  }
  const float* xb = x + (size_t)b*CCH*NPIX;
  float* kb = kimg + (size_t)b*CCH*NPIX;
  for (int c = 0; c < CCH; ++c){
    const float* pl = xb + (size_t)c*NPIX;
    float vsum = wgt[0]*pl[idx[0]] + wgt[1]*pl[idx[1]] + wgt[2]*pl[idx[2]] + wgt[3]*pl[idx[3]];
    kb[(size_t)c*NPIX + pix] = pl[pix] + vsum;
  }
}

// ---------------- gather kept tokens -> bf16 matrices.
// Qin/Kin row-major [m][c]; V1t transposed per 64-key tile: V1t[(tile*192+c)*64 + key]
// so k_attn can stage and fragment-load V with pure b128 LDS ops.
__global__ void __launch_bounds__(256)
k_gather(const float* __restrict__ x, const float* __restrict__ kimg,
         const float* __restrict__ v, const int* __restrict__ idx1,
         __bf16* __restrict__ Qin, __bf16* __restrict__ Kin, __bf16* __restrict__ V1t){
  int i = blockIdx.x*256 + threadIdx.x;
  int b = i / NKEEP, m = i % NKEEP;
  int tok = idx1[b*NKEEP + m];
  size_t rb = ((size_t)b*NKEEP + m)*CCH;
  size_t xb = (size_t)b*CCH*NPIX + tok;
  size_t vt = (((size_t)b*(NKEEP/64) + (m >> 6))*CCH)*64 + (m & 63);
  for (int c = 0; c < CCH; ++c){
    Qin[rb + c]            = (__bf16)x[xb + (size_t)c*NPIX];
    Kin[rb + c]            = (__bf16)kimg[xb + (size_t)c*NPIX];
    V1t[vt + (size_t)c*64] = (__bf16)v[xb + (size_t)c*NPIX];
  }
}

// ---------------- flash attention over kept tokens.
// S^T = K.Q^T so softmax stats are per-lane; S^T D-frags repack into the P A-frag
// with no cross-lane shuffles. gridDim.z splits the 192 value channels in half
// (each block recomputes identical S^T/stats) to keep the O accumulator at 6
// D-tiles (48 VGPRs) and avoid scratch spills. Q tile is loop-invariant -> LDS.
// V arrives pre-transposed (channel-major per 64-key tile) so all LDS traffic in
// the hot loop is b128-vectorized.
__global__ void __launch_bounds__(128, 1)
k_attn(const __bf16* __restrict__ q1, const __bf16* __restrict__ k1,
       const __bf16* __restrict__ v1t, const int* __restrict__ idx1,
       float* __restrict__ out_img){
  __shared__ __align__(16) __bf16 qL[64*CCH];       // [query][c]
  __shared__ __align__(16) __bf16 kL[64*CCH];       // [key][c]
  __shared__ __align__(16) __bf16 vL[CCH*64];       // [c][key]  (transposed)
  __shared__ int tokL[64];
  const int t = threadIdx.x, lane = t & 31, wave = t >> 5;
  const int lo = lane & 15, hi = lane >> 4;
  const int b = blockIdx.y, qblk = blockIdx.x*64;
  const int c0 = blockIdx.z*96;               // channel half base
  if (t < 64) tokL[t] = idx1[b*NKEEP + qblk + t];
  { // stage Q tile once (loop-invariant)
    const uint4* Qs = (const uint4*)(q1 + ((size_t)b*NKEEP + qblk)*CCH);
    uint4* qd = (uint4*)qL;
    #pragma unroll
    for (int i = 0; i < 12; ++i){ int cid = t + 128*i; qd[cid] = Qs[cid]; }
  }
  v8f o[6] = {};
  float mi = -1e30f, li = 0.f;
  const float scale = 0.07216878364870322f; // 192^-0.5
  const int qrow = wave*16 + lo;
  for (int kb = 0; kb < NKEEP; kb += 64){
    const uint4* Ks = (const uint4*)(k1 + ((size_t)b*NKEEP + kb)*CCH);
    const uint4* Vs = (const uint4*)(v1t + (((size_t)b*(NKEEP/64) + (kb >> 6))*CCH)*64);
    uint4* kd = (uint4*)kL; uint4* vd = (uint4*)vL;
    #pragma unroll
    for (int i = 0; i < 12; ++i){ int cid = t + 128*i; kd[cid] = Ks[cid]; vd[cid] = Vs[cid]; }
    __syncthreads();
    v8f st[4] = {};
    #pragma unroll
    for (int ch = 0; ch < 6; ++ch){
      v16bf qf;   // B-frag: lane lo = query, elems = 16 contiguous k
      #pragma unroll
      for (int e = 0; e < 16; ++e)
        qf[e] = qL[qrow*CCH + ch*32 + 16*hi + e];
      #pragma unroll
      for (int tt = 0; tt < 4; ++tt){
        v16bf af; // A-frag: lane lo = key row
        #pragma unroll
        for (int e = 0; e < 16; ++e){
          int k = ch*32 + 16*hi + e;
          af[e] = kL[(tt*16 + lo)*CCH + k];
        }
        st[tt] = __builtin_amdgcn_wmma_f32_16x16x32_bf16(false, af, false, qf, (short)0, st[tt], false, false);
      }
    }
    float cm = -1e30f;
    #pragma unroll
    for (int tt = 0; tt < 4; ++tt)
      #pragma unroll
      for (int r = 0; r < 8; ++r){ st[tt][r] *= scale; cm = fmaxf(cm, st[tt][r]); }
    cm = fmaxf(cm, __shfl_xor(cm, 16, 32));
    float mnew = fmaxf(mi, cm);
    float alpha = __expf(mi - mnew);
    float csum = 0.f;
    #pragma unroll
    for (int tt = 0; tt < 4; ++tt)
      #pragma unroll
      for (int r = 0; r < 8; ++r){ float pp = __expf(st[tt][r] - mnew); st[tt][r] = pp; csum += pp; }
    csum += __shfl_xor(csum, 16, 32);
    li = li*alpha + csum; mi = mnew;
    float av[8];
    #pragma unroll
    for (int r = 0; r < 8; ++r) av[r] = __shfl(alpha, hi*8 + r, 32);
    #pragma unroll
    for (int ct = 0; ct < 6; ++ct)
      #pragma unroll
      for (int r = 0; r < 8; ++r) o[ct][r] *= av[r];
    // repack P^T D-frags -> P A-frags (same lane, pure element shuffle)
    v16bf pa[2];
    #pragma unroll
    for (int kc = 0; kc < 2; ++kc)
      #pragma unroll
      for (int e = 0; e < 16; ++e)
        pa[kc][e] = (__bf16)st[2*kc + (e >> 3)][e & 7];
    #pragma unroll
    for (int ct = 0; ct < 6; ++ct){
      #pragma unroll
      for (int kc = 0; kc < 2; ++kc){
        v16bf bfv; // B-frag: lane lo = channel col; elems = keys in table order (8hi+s, 16+8hi+s)
        #pragma unroll
        for (int e = 0; e < 16; ++e)
          bfv[e] = vL[(c0 + ct*16 + lo)*64 + kc*32 + (e & 7) + 8*hi + ((e >> 3) << 4)];
        o[ct] = __builtin_amdgcn_wmma_f32_16x16x32_bf16(false, pa[kc], false, bfv, (short)0, o[ct], false, false);
      }
    }
    __syncthreads();
  }
  float lv[8];
  #pragma unroll
  for (int r = 0; r < 8; ++r) lv[r] = __shfl(li, hi*8 + r, 32);
  int toks[8];
  #pragma unroll
  for (int r = 0; r < 8; ++r) toks[r] = tokL[wave*16 + r + 8*hi];
  float* ob = out_img + (size_t)b*CCH*NPIX;
  #pragma unroll
  for (int ct = 0; ct < 6; ++ct)
    #pragma unroll
    for (int r = 0; r < 8; ++r)
      ob[(size_t)(c0 + ct*16 + lo)*NPIX + toks[r]] = o[ct][r] / lv[r];
}

// ---------------- non-kept tokens get v*sa
__global__ void __launch_bounds__(256)
k_scatter2(const float* __restrict__ v, const float* __restrict__ sa,
           const int* __restrict__ idx2, float* __restrict__ out_img){
  int i = blockIdx.x*256 + threadIdx.x;
  int b = i / NKEEP, m = i % NKEEP;
  int tok = idx2[b*NKEEP + m];
  float s = sa[(size_t)b*NPIX + tok];
  const float* vb = v + (size_t)b*CCH*NPIX + tok;
  float* ob = out_img + (size_t)b*CCH*NPIX + tok;
  for (int c = 0; c < CCH; ++c)
    ob[(size_t)c*NPIX] = vb[(size_t)c*NPIX] * s;
}

// ---------------- depthwise 3x3 + gelu*ca + residual
__global__ void __launch_bounds__(256)
k_dw(const float* __restrict__ out_img, const float* __restrict__ cw,
     const float* __restrict__ cb, const float* __restrict__ ca,
     float* __restrict__ out2){
  int i = blockIdx.x*256 + threadIdx.x;
  int pix = i % NPIX; int c = (i / NPIX) % CCH; int b = i / (NPIX*CCH);
  int h = pix / WW2, w = pix % WW2;
  const float* src = out_img + ((size_t)b*CCH + c)*NPIX;
  float acc = cb[c];
  #pragma unroll
  for (int dy = -1; dy <= 1; ++dy)
    #pragma unroll
    for (int dx = -1; dx <= 1; ++dx){
      int hy = h+dy, wx = w+dx;
      if (hy < 0 || hy >= HH || wx < 0 || wx >= WW2) continue;
      acc += cw[c*9 + (dy+1)*3 + (dx+1)] * src[hy*WW2 + wx];
    }
  out2[((size_t)b*CCH + c)*NPIX + pix] = gelu_t(acc) * ca[b*CCH + c] + src[pix];
}

extern "C" void kernel_launch(void* const* d_in, const int* in_sizes, int n_in,
                              void* d_out, int out_size, void* d_ws, size_t ws_size,
                              hipStream_t stream)
{
  const float* x        = (const float*)d_in[0];
  const float* p_in_w   = (const float*)d_in[1];
  const float* p_in_b   = (const float*)d_in[2];
  const float* p_ln_w   = (const float*)d_in[3];
  const float* p_ln_b   = (const float*)d_in[4];
  const float* p_off_w1 = (const float*)d_in[5];
  const float* p_off_b1 = (const float*)d_in[6];
  const float* p_off_w2 = (const float*)d_in[7];
  const float* p_off_b2 = (const float*)d_in[8];
  const float* p_ca_w   = (const float*)d_in[9];
  const float* p_ca_b   = (const float*)d_in[10];
  const float* p_sa_w   = (const float*)d_in[11];
  const float* p_sa_b   = (const float*)d_in[12];
  const float* p_mask_w1= (const float*)d_in[13];
  const float* p_mask_b1= (const float*)d_in[14];
  const float* p_mask_w2= (const float*)d_in[15];
  const float* p_mask_b2= (const float*)d_in[16];
  const float* v_w      = (const float*)d_in[17];
  const float* v_b      = (const float*)d_in[18];
  const float* q_w      = (const float*)d_in[19];
  const float* q_b      = (const float*)d_in[20];
  const float* k_w      = (const float*)d_in[21];
  const float* k_b      = (const float*)d_in[22];
  const float* cs_w     = (const float*)d_in[23];
  const float* cs_b     = (const float*)d_in[24];
  const float* out_w    = (const float*)d_in[25];
  const float* out_b    = (const float*)d_in[26];
  (void)in_sizes; (void)n_in; (void)out_size; (void)ws_size;

  char* ws = (char*)d_ws; size_t off = 0;
  auto alloc = [&](size_t bytes)->void*{
    void* p = ws + off; off = (off + bytes + 255) & ~(size_t)255; return p;
  };
  float*  f       = (float*)alloc((size_t)BATCH*NC4*NPIX*4);
  float*  offsets = (float*)alloc((size_t)BATCH*2*NPIX*4);
  float*  sa      = (float*)alloc((size_t)BATCH*NPIX*4);
  float*  score   = (float*)alloc((size_t)BATCH*NPIX*4);
  float*  fmean   = (float*)alloc((size_t)BATCH*NC4*4);
  float*  ca      = (float*)alloc((size_t)BATCH*CCH*4);
  int*    idx1    = (int*)alloc((size_t)BATCH*NKEEP*4);
  int*    idx2    = (int*)alloc((size_t)BATCH*NKEEP*4);
  float*  v       = (float*)alloc((size_t)BATCH*CCH*NPIX*4);
  float*  kimg    = (float*)alloc((size_t)BATCH*CCH*NPIX*4);
  __bf16* Qin     = (__bf16*)alloc((size_t)BATCH*NKEEP*CCH*2);
  __bf16* Kin     = (__bf16*)alloc((size_t)BATCH*NKEEP*CCH*2);
  __bf16* V1t     = (__bf16*)alloc((size_t)BATCH*NKEEP*CCH*2);
  __bf16* Q1      = (__bf16*)alloc((size_t)BATCH*NKEEP*CCH*2);
  __bf16* K1      = (__bf16*)alloc((size_t)BATCH*NKEEP*CCH*2);
  float*  out_img = (float*)alloc((size_t)BATCH*CCH*NPIX*4);
  float*  out2    = (float*)alloc((size_t)BATCH*CCH*NPIX*4);

  k_f<<<BATCH*NPIX/256, 256, 0, stream>>>(x, p_in_w, p_in_b, p_ln_w, p_ln_b, f);
  k_off_score<<<BATCH*NPIX/256, 256, 0, stream>>>(f, p_off_w1, p_off_b1, p_off_w2, p_off_b2,
                                                  p_mask_w1, p_mask_b1, p_mask_w2, p_mask_b2,
                                                  offsets, score);
  k_sa<<<BATCH*NPIX/256, 256, 0, stream>>>(f, p_sa_w, p_sa_b, sa);
  k_fmean<<<dim3(NC4, BATCH), 256, 0, stream>>>(f, fmean);
  k_ca<<<(BATCH*CCH + 255)/256, 256, 0, stream>>>(fmean, p_ca_w, p_ca_b, ca);
  k_rank<<<BATCH*NPIX/256, 256, 0, stream>>>(score, idx1, idx2);
  k_gemm<0,0><<<dim3(NPIX/64, CCH/64, BATCH), 256, 0, stream>>>(
      x, NPIX, (long)CCH*NPIX, v_w, CCH, v_b, v, NPIX, (long)CCH*NPIX);
  k_warp<<<BATCH*NPIX/256, 256, 0, stream>>>(x, offsets, kimg);
  k_gather<<<BATCH*NKEEP/256, 256, 0, stream>>>(x, kimg, v, idx1, Qin, Kin, V1t);
  k_gemm<1,1><<<dim3(NKEEP/64, CCH/64, BATCH), 256, 0, stream>>>(
      Qin, CCH, (long)NKEEP*CCH, q_w, CCH, q_b, Q1, CCH, (long)NKEEP*CCH);
  k_gemm<1,1><<<dim3(NKEEP/64, CCH/64, BATCH), 256, 0, stream>>>(
      Kin, CCH, (long)NKEEP*CCH, k_w, CCH, k_b, K1, CCH, (long)NKEEP*CCH);
  k_attn<<<dim3(NKEEP/64, BATCH, 2), 128, 0, stream>>>(Q1, K1, V1t, idx1, out_img);
  k_scatter2<<<BATCH*NKEEP/256, 256, 0, stream>>>(v, sa, idx2, out_img);
  k_dw<<<(BATCH*CCH*NPIX)/256, 256, 0, stream>>>(out_img, cs_w, cs_b, ca, out2);
  k_gemm<0,0><<<dim3(NPIX/64, CCH/64, BATCH), 256, 0, stream>>>(
      out2, NPIX, (long)CCH*NPIX, out_w, CCH, out_b, (float*)d_out, NPIX, (long)CCH*NPIX);
}
